// Tree_SSM_25795573580018
// MI455X (gfx1250) — matrix operand order repaired and checked by reading, not compiled, not run on hardware
//
#include <hip/hip_runtime.h>
#include <hip/hip_bf16.h>

// ---------------------------------------------------------------------------
// Tree-SSM pipeline for MI455X (gfx1250, wave32).
//   Stage 1: in_proj GEMM (bf16 WMMA), silu on z half
//   Stage 2: depthwise 3x3 conv + silu
//   Stage 3: x_dbl (14-way dot, wave-shuffle reduce) ; dts/dA/dBx elementwise
//   Stage 4: tree scan: 12 pointer-jumping rounds (ping-pong buffers)
//   Stage 5: LN(h)*Csv + Ds*xs -> LN -> *z -> bf16
//   Stage 6: out_proj GEMM (bf16 WMMA) -> fp32 out
// All layouts are (b, l, e) with e innermost.
// ---------------------------------------------------------------------------

#define Bn   4
#define Hn   64
#define Wn   64
#define DMn  192
#define DEn  384
#define Rn   12
#define Ln   (Hn * Wn)          // 4096
#define BL   (Bn * Ln)          // 16384
#define N1   (2 * DEn)          // 768
#define SCAN_ITERS 12           // ceil(log2(4096))

typedef __attribute__((ext_vector_type(16))) __bf16        v16bf;
typedef __attribute__((ext_vector_type(8)))  float         v8f;
typedef __attribute__((ext_vector_type(4)))  unsigned int  u32x4;

union V16 { v16bf v; u32x4 q[2]; };

__device__ __forceinline__ unsigned short f2bf(float f) {
    unsigned u = __float_as_uint(f);
    unsigned r = u + 0x7FFFu + ((u >> 16) & 1u);   // round-to-nearest-even
    return (unsigned short)(r >> 16);
}
__device__ __forceinline__ float silu(float v) {
    return v / (1.0f + __expf(-v));
}

// --------------------------- fp32 -> bf16 ----------------------------------
__global__ void k_f32_to_bf16(const float* __restrict__ in,
                              unsigned short* __restrict__ out, int n) {
    int i = blockIdx.x * blockDim.x + threadIdx.x;
    if (i < n) out[i] = f2bf(in[i]);
}

// --------------------------- WMMA GEMM -------------------------------------
// C[M,N] = A[M,K](bf16) * Bw[N,K](bf16)^T, f32 accumulate.
// One wave -> 16(M) x 64(N) tile, K stepped by 32 (v_wmma_f32_16x16x32_bf16).
// Operand fetch uses contiguous k order, same for A and B:
//   A lane l: row m0+(l&15), halves  k+(l>=16)*16 .. +15  (contiguous 32B)
//   B lane l: col n0+(l&15), halves  k+(l>=16)*16 .. +15  (contiguous 32B)
// mode 0: split epilogue -> o0 = x1 (cols<384), o1 = silu -> z (cols>=384)
// mode 1: plain f32 store to o0 with ld = N
__global__ void k_wmma_gemm(const unsigned short* __restrict__ A,
                            const unsigned short* __restrict__ Bw,
                            float* __restrict__ o0, float* __restrict__ o1,
                            int M, int N, int K, int mode) {
    int wid  = (blockIdx.x * blockDim.x + threadIdx.x) >> 5;
    int lane = threadIdx.x & 31;
    int nt   = N >> 6;
    int mIdx = wid / nt;
    int nIdx = wid - mIdx * nt;
    if (mIdx >= (M >> 4)) return;
    int m0 = mIdx << 4, n0 = nIdx << 6;
    int hi2 = lane >> 4;       // lane-half select
    int l15 = lane & 15;

    const unsigned short* aptr = A + (size_t)(m0 + l15) * K + hi2 * 16;
    v8f acc[4] = {};

    for (int k = 0; k < K; k += 32) {
        V16 a;
        a.q[0] = *(const u32x4*)(aptr + k);
        a.q[1] = *(const u32x4*)(aptr + k + 8);
        __builtin_prefetch(aptr + k + 32);   // global_prefetch_b8 next k-slab
#pragma unroll
        for (int s = 0; s < 4; ++s) {
            V16 b;
            const unsigned short* bp =
                Bw + (size_t)(n0 + s * 16 + l15) * K + k + hi2 * 16;
            b.q[0] = *(const u32x4*)(bp);
            b.q[1] = *(const u32x4*)(bp + 8);
            acc[s] = __builtin_amdgcn_wmma_f32_16x16x32_bf16(
                false, a.v, false, b.v, (short)0, acc[s], false, false);
        }
    }

    int cr = hi2 * 8;          // C/D layout: vgpr i -> row i (+8 for lanes 16..31)
#pragma unroll
    for (int s = 0; s < 4; ++s) {
#pragma unroll
        for (int i = 0; i < 8; ++i) {
            int row = m0 + cr + i;
            int col = n0 + s * 16 + l15;
            float vv = acc[s][i];
            if (mode == 0) {
                if (col < DEn) o0[(size_t)row * DEn + col] = vv;
                else           o1[(size_t)row * DEn + (col - DEn)] = silu(vv);
            } else {
                o0[(size_t)row * N + col] = vv;
            }
        }
    }
}

// ----------------------- depthwise 3x3 conv + silu -------------------------
__global__ void k_dwconv(const float* __restrict__ x1,
                         const float* __restrict__ cw,
                         const float* __restrict__ cb,
                         float* __restrict__ xs) {
    int idx = blockIdx.x * blockDim.x + threadIdx.x;
    if (idx >= BL * DEn) return;
    int e = idx % DEn;
    int p = idx / DEn;
    int w = p % Wn;
    int h = (p / Wn) % Hn;
    int b = p / (Wn * Hn);
    float s = cb[e];
#pragma unroll
    for (int dy = -1; dy <= 1; ++dy) {
#pragma unroll
        for (int dx = -1; dx <= 1; ++dx) {
            int hh = h + dy, ww = w + dx;
            if (hh >= 0 && hh < Hn && ww >= 0 && ww < Wn) {
                s += x1[((size_t)(b * Ln + hh * Wn + ww)) * DEn + e] *
                     cw[e * 9 + (dy + 1) * 3 + (dx + 1)];
            }
        }
    }
    xs[idx] = silu(s);
}

// ----------------------- x_dbl: 14 dots of length 384 ----------------------
__global__ void k_xdbl(const float* __restrict__ xs,
                       const float* __restrict__ xpw,
                       float* __restrict__ xdbl) {
    __shared__ float w[(Rn + 2) * DEn];
    for (int i = threadIdx.x; i < (Rn + 2) * DEn; i += blockDim.x) w[i] = xpw[i];
    __syncthreads();
    int wv = threadIdx.x >> 5, lane = threadIdx.x & 31;
    int pos = blockIdx.x * 8 + wv;
    if (pos >= BL) return;
    const float* xr = xs + (size_t)pos * DEn;
    float acc[Rn + 2];
#pragma unroll
    for (int c = 0; c < Rn + 2; ++c) acc[c] = 0.0f;
    for (int d = lane; d < DEn; d += 32) {
        float xv = xr[d];
#pragma unroll
        for (int c = 0; c < Rn + 2; ++c) acc[c] += w[c * DEn + d] * xv;
    }
#pragma unroll
    for (int off = 16; off > 0; off >>= 1)
#pragma unroll
        for (int c = 0; c < Rn + 2; ++c) acc[c] += __shfl_down(acc[c], off, 32);
    if (lane == 0) {
        float* o = xdbl + (size_t)pos * 16;
#pragma unroll
        for (int c = 0; c < Rn + 2; ++c) o[c] = acc[c];
        o[14] = 0.0f; o[15] = 0.0f;
    }
}

// -------------- dts = softplus(W_dt * x_dbl[:12] + b) ; dA ; dBx -----------
__global__ void k_dts(const float* __restrict__ xs,
                      const float* __restrict__ xdbl,
                      const float* __restrict__ dtw,
                      const float* __restrict__ dtb,
                      const float* __restrict__ A_logs,
                      float* __restrict__ dA, float* __restrict__ dBx) {
    int i = blockIdx.x * blockDim.x + threadIdx.x;
    if (i >= BL * DEn) return;
    int pos = i / DEn, e = i - pos * DEn;
    const float* xd = xdbl + (size_t)pos * 16;
    float t = dtb[e];
#pragma unroll
    for (int r = 0; r < Rn; ++r) t += dtw[e * Rn + r] * xd[r];
    float sp = (t > 20.0f) ? t : log1pf(__expf(t));
    float negA = __expf(A_logs[e]);
    dA[i]  = __expf(sp * negA);
    dBx[i] = sp * xd[Rn] * xs[i];      // xd[12] = Bsv
}

// ----------------------- tree scan init (gather by sorted_index) -----------
__global__ void k_scan_init(const float* __restrict__ dBx,
                            const float* __restrict__ dA,
                            const int* __restrict__ si,
                            const int* __restrict__ sp,
                            float* __restrict__ v0, float* __restrict__ c0,
                            int* __restrict__ ptr0, int* __restrict__ inv) {
    int i = blockIdx.x * blockDim.x + threadIdx.x;
    if (i >= BL * DEn) return;
    int pos = i / DEn, e = i - pos * DEn;
    int b = pos / Ln, l = pos - b * Ln;
    int s = si[pos];
    size_t src = ((size_t)(b * Ln + s)) * DEn + e;
    v0[i] = dBx[src];
    c0[i] = dA[src];
    if (e == 0) {
        ptr0[pos] = sp[pos];
        inv[b * Ln + s] = l;
    }
}

// -------------------- one pointer-jumping round (float4) -------------------
__global__ void k_scan_iter(const float4* __restrict__ vin,
                            const float4* __restrict__ cin,
                            const int* __restrict__ pin,
                            float4* __restrict__ vout,
                            float4* __restrict__ cout,
                            int* __restrict__ pout) {
    const int QW = DEn / 4;                         // 96
    int i = blockIdx.x * blockDim.x + threadIdx.x;
    if (i >= BL * QW) return;
    int pos = i / QW, q = i - pos * QW;
    int b = pos / Ln;
    int m = pin[pos];
    float4 v = vin[(size_t)pos * QW + q];
    float4 c = cin[(size_t)pos * QW + q];
    if (m >= 0) {
        size_t pr = ((size_t)(b * Ln + m)) * QW + q;
        float4 vp = vin[pr], cp = cin[pr];
        v.x += c.x * vp.x; v.y += c.y * vp.y;
        v.z += c.z * vp.z; v.w += c.w * vp.w;
        c.x *= cp.x; c.y *= cp.y; c.z *= cp.z; c.w *= cp.w;
    } else {
        c = make_float4(0.f, 0.f, 0.f, 0.f);        // matches reference where()
    }
    vout[(size_t)pos * QW + q] = v;
    cout[(size_t)pos * QW + q] = c;
    if (q == 0) pout[pos] = (m >= 0) ? pin[b * Ln + m] : -1;
}

// ------------- LN(h)*Csv + Ds*xs -> LN -> *z -> bf16  (block/pos) ----------
__device__ __forceinline__ void blockReduce2(float& a, float& b,
                                             float* s1, float* s2) {
    int t = threadIdx.x;
    s1[t] = a; s2[t] = b; __syncthreads();
#pragma unroll
    for (int off = 64; off > 0; off >>= 1) {
        if (t < off) { s1[t] += s1[t + off]; s2[t] += s2[t + off]; }
        __syncthreads();
    }
    a = s1[0]; b = s2[0]; __syncthreads();
}

__global__ void k_norm(const float* __restrict__ val,
                       const int* __restrict__ inv,
                       const float* __restrict__ xdbl,
                       const float* __restrict__ xs,
                       const float* __restrict__ hg, const float* __restrict__ hb,
                       const float* __restrict__ Ds,
                       const float* __restrict__ og, const float* __restrict__ ob,
                       const float* __restrict__ z,
                       unsigned short* __restrict__ ybf) {
    __shared__ float s1[128], s2[128];
    int pos = blockIdx.x;
    int b = pos / Ln;
    int t = threadIdx.x;
    const float* hr = val + ((size_t)(b * Ln + inv[pos])) * DEn;
    float h[3], y[3];
    float sum = 0.f, ss = 0.f;
#pragma unroll
    for (int j = 0; j < 3; ++j) {
        h[j] = hr[t + 128 * j];
        sum += h[j]; ss += h[j] * h[j];
    }
    blockReduce2(sum, ss, s1, s2);
    float mu = sum * (1.0f / DEn);
    float rs = rsqrtf(ss * (1.0f / DEn) - mu * mu + 1e-5f);
    float Csv = xdbl[(size_t)pos * 16 + 13];
    sum = 0.f; ss = 0.f;
#pragma unroll
    for (int j = 0; j < 3; ++j) {
        int e = t + 128 * j;
        float hn = (h[j] - mu) * rs * hg[e] + hb[e];
        y[j] = hn * Csv + Ds[e] * xs[(size_t)pos * DEn + e];
        sum += y[j]; ss += y[j] * y[j];
    }
    blockReduce2(sum, ss, s1, s2);
    float mu2 = sum * (1.0f / DEn);
    float rs2 = rsqrtf(ss * (1.0f / DEn) - mu2 * mu2 + 1e-5f);
#pragma unroll
    for (int j = 0; j < 3; ++j) {
        int e = t + 128 * j;
        float yo = (y[j] - mu2) * rs2 * og[e] + ob[e];
        yo *= z[(size_t)pos * DEn + e];
        ybf[(size_t)pos * DEn + e] = f2bf(yo);
    }
}

// ---------------------------------------------------------------------------
extern "C" void kernel_launch(void* const* d_in, const int* in_sizes, int n_in,
                              void* d_out, int out_size, void* d_ws, size_t ws_size,
                              hipStream_t stream) {
    const float* x       = (const float*)d_in[0];
    const float* inW     = (const float*)d_in[1];
    const float* convW   = (const float*)d_in[2];
    const float* convB   = (const float*)d_in[3];
    const float* xpw     = (const float*)d_in[4];
    const float* dtw     = (const float*)d_in[5];
    const float* dtb     = (const float*)d_in[6];
    const float* A_logs  = (const float*)d_in[7];
    const float* Ds      = (const float*)d_in[8];
    const float* hg      = (const float*)d_in[9];
    const float* hb      = (const float*)d_in[10];
    const float* og      = (const float*)d_in[11];
    const float* ob      = (const float*)d_in[12];
    const float* outW    = (const float*)d_in[13];
    const int*   si      = (const int*)d_in[14];
    const int*   sp      = (const int*)d_in[15];

    char* ws = (char*)d_ws;
    size_t cur = 0;
    auto alloc = [&](size_t bytes) -> char* {
        char* p = ws + cur;
        cur += (bytes + 255) & ~(size_t)255;
        return p;
    };
    const size_t SZ = (size_t)BL * DEn * sizeof(float);   // 25.2 MB
    float* zbuf = (float*)alloc(SZ);
    float* x1   = (float*)alloc(SZ);
    float* xs   = (float*)alloc(SZ);
    float* dAb  = (float*)alloc(SZ);
    float* dBxb = (float*)alloc(SZ);
    float* V0   = (float*)alloc(SZ);
    float* C0   = (float*)alloc(SZ);
    float* V1   = (float*)alloc(SZ);
    float* C1   = (float*)alloc(SZ);
    float* xdbl = (float*)alloc((size_t)BL * 16 * sizeof(float));
    int*   ptr0 = (int*)alloc((size_t)BL * sizeof(int));
    int*   ptr1 = (int*)alloc((size_t)BL * sizeof(int));
    int*   inv  = (int*)alloc((size_t)BL * sizeof(int));
    unsigned short* xb  = (unsigned short*)alloc((size_t)BL * DMn * 2);
    unsigned short* wb1 = (unsigned short*)alloc((size_t)N1 * DMn * 2);
    unsigned short* wb2 = (unsigned short*)alloc((size_t)DMn * DEn * 2);
    unsigned short* ybf = (unsigned short*)alloc((size_t)BL * DEn * 2);

    // Stage 0: bf16 conversions for the WMMA operands
    {
        int n = BL * DMn;
        k_f32_to_bf16<<<(n + 255) / 256, 256, 0, stream>>>(x, xb, n);
        n = N1 * DMn;
        k_f32_to_bf16<<<(n + 255) / 256, 256, 0, stream>>>(inW, wb1, n);
        n = DMn * DEn;
        k_f32_to_bf16<<<(n + 255) / 256, 256, 0, stream>>>(outW, wb2, n);
    }

    // Stage 1: in_proj GEMM (16384 x 192) * (192 -> 768), split + silu(z)
    {
        int waves = (BL / 16) * (N1 / 64);            // 12288
        k_wmma_gemm<<<(waves + 7) / 8, 256, 0, stream>>>(
            xb, wb1, x1, zbuf, BL, N1, DMn, 0);
    }

    // Stage 2: depthwise 3x3 conv + silu
    {
        int n = BL * DEn;
        k_dwconv<<<(n + 255) / 256, 256, 0, stream>>>(x1, convW, convB, xs);
    }

    // Stage 3: x_dbl then dts/dA/dBx
    k_xdbl<<<BL / 8, 256, 0, stream>>>(xs, xpw, xdbl);
    {
        int n = BL * DEn;
        k_dts<<<(n + 255) / 256, 256, 0, stream>>>(
            xs, xdbl, dtw, dtb, A_logs, dAb, dBxb);
    }

    // Stage 4: tree scan (pointer jumping, 12 rounds, ping-pong)
    {
        int n = BL * DEn;
        k_scan_init<<<(n + 255) / 256, 256, 0, stream>>>(
            dBxb, dAb, si, sp, V0, C0, ptr0, inv);
        float *va = V0, *vb = V1, *ca = C0, *cb = C1;
        int *pa = ptr0, *pb = ptr1;
        int nq = BL * (DEn / 4);
        for (int it = 0; it < SCAN_ITERS; ++it) {
            k_scan_iter<<<(nq + 255) / 256, 256, 0, stream>>>(
                (const float4*)va, (const float4*)ca, pa,
                (float4*)vb, (float4*)cb, pb);
            float* tf;
            tf = va; va = vb; vb = tf;
            tf = ca; ca = cb; cb = tf;
            int* ti = pa; pa = pb; pb = ti;
        }
        // SCAN_ITERS even -> result back in V0 (== va)
        k_norm<<<BL, 128, 0, stream>>>(
            va, inv, xdbl, xs, hg, hb, Ds, og, ob, zbuf, ybf);
    }

    // Stage 6: out_proj GEMM (16384 x 384) * (384 -> 192) -> fp32 out
    {
        int waves = (BL / 16) * (DMn / 64);           // 3072
        k_wmma_gemm<<<(waves + 7) / 8, 256, 0, stream>>>(
            ybf, wb2, (float*)d_out, nullptr, BL, DMn, DEn, 1);
    }
}
